// LightweightAttnBlock_86560770883948
// MI455X (gfx1250) — matrix operand order repaired and checked by reading, not compiled, not run on hardware
//
#include <hip/hip_runtime.h>

typedef __bf16 bf16;
typedef __attribute__((ext_vector_type(16))) __bf16 v16bf;
typedef __attribute__((ext_vector_type(8)))  __bf16 v8bf;
typedef __attribute__((ext_vector_type(4)))  __bf16 v4bf;
typedef __attribute__((ext_vector_type(2)))  __bf16 v2bf;
typedef __attribute__((ext_vector_type(8)))  float  v8f;

#define B_DIM   4
#define C_DIM   256
#define N_PIX   4096
#define HEADS   8
#define HD      32
#define GROUPS  16

union FragAB {            // build v16bf from two contiguous 16B halves
    v16bf v;
    struct { v8bf lo, hi; } p;
};

// ---------------------------------------------------------------------------
// fp32 -> bf16 weight conversion
// ---------------------------------------------------------------------------
__global__ void cvt_w_kernel(const float* __restrict__ src, bf16* __restrict__ dst, int n) {
    int i = blockIdx.x * 256 + threadIdx.x;
    if (i < n) dst[i] = (bf16)src[i];
}

// ---------------------------------------------------------------------------
// GroupNorm: one block per (b, group). Writes bf16 h_[b][c][n]. float4 I/O.
// ---------------------------------------------------------------------------
__global__ void groupnorm_kernel(const float* __restrict__ x,
                                 const float* __restrict__ gw,
                                 const float* __restrict__ gb,
                                 bf16* __restrict__ hout) {
    const int CPG = C_DIM / GROUPS;          // 16
    const int M   = CPG * N_PIX;             // 65536 elements per group
    const int M4  = M / 4;
    int b = blockIdx.x / GROUPS;
    int g = blockIdx.x % GROUPS;
    const float*  xg  = x + (size_t)(b * C_DIM + g * CPG) * N_PIX;
    const float4* xg4 = (const float4*)xg;

    float s = 0.f, ss = 0.f;
    for (int t = threadIdx.x; t < M4; t += 256) {
        float4 v = xg4[t];
        s  += v.x + v.y + v.z + v.w;
        ss += v.x * v.x + v.y * v.y + v.z * v.z + v.w * v.w;
    }
    __shared__ float rs[256], rss[256];
    rs[threadIdx.x] = s; rss[threadIdx.x] = ss;
    __syncthreads();
    for (int off = 128; off > 0; off >>= 1) {
        if ((int)threadIdx.x < off) {
            rs[threadIdx.x]  += rs[threadIdx.x + off];
            rss[threadIdx.x] += rss[threadIdx.x + off];
        }
        __syncthreads();
    }
    float mean = rs[0] / (float)M;
    float var  = rss[0] / (float)M - mean * mean;
    float rstd = rsqrtf(var + 1e-6f);

    bf16* hg = hout + (size_t)(b * C_DIM + g * CPG) * N_PIX;
    for (int t = threadIdx.x; t < M4; t += 256) {
        int c = g * CPG + (t * 4) / N_PIX;   // 4096 % 4 == 0: one c per float4
        float sc = rstd * gw[c], bi = gb[c] - mean * sc;
        float4 v = xg4[t];
        v4bf o;
        o[0] = (bf16)(v.x * sc + bi);
        o[1] = (bf16)(v.y * sc + bi);
        o[2] = (bf16)(v.z * sc + bi);
        o[3] = (bf16)(v.w * sc + bi);
        *(v4bf*)&hg[t * 4] = o;
    }
}

// ---------------------------------------------------------------------------
// 1x1-conv GEMM: out[b][co][n] = sum_ci W[co][ci] * In[b][ci][n] + bias[co]
// Block: 256 thr = 8 waves; block tile 128(co) x 32(n); per wave 16x32.
// B panel staged TRANSPOSED in LDS -> all fragment reads are b128.
// mode 1: bf16 out [b][h][n][d]                  (k / v for attention)
// mode 3: bf16 out [b][h][n][d], value * scale   (q, softmax scale folded)
// mode 2: f32  out = val + resid                 (final proj + residual)
// ---------------------------------------------------------------------------
__global__ void gemm_kernel(const bf16* __restrict__ W,
                            const bf16* __restrict__ In,
                            const float* __restrict__ bias,
                            bf16* __restrict__ outb,
                            float* __restrict__ outf,
                            const float* __restrict__ resid,
                            int mode) {
    const int NT = N_PIX / 32;               // 128 n-tiles
    int bid = blockIdx.x;
    int b   = bid / (2 * NT);
    int r2  = bid % (2 * NT);
    int cb  = r2 / NT;                       // 0..1 : 128-row co block
    int n0  = (r2 % NT) * 32;

    int lane = threadIdx.x & 31;
    int wv   = threadIdx.x >> 5;             // 0..7
    int co0  = cb * 128 + wv * 16;
    int hf   = lane >> 4;                    // half-wave
    int l16  = lane & 15;

    // ---- stage B panel transposed: BsT[n][ci], padded row (264) for banks
    __shared__ bf16 BsT[32][264];
    {
        const bf16* inrow = In + ((size_t)b * C_DIM + threadIdx.x) * N_PIX + n0;
#pragma unroll
        for (int c4 = 0; c4 < 4; ++c4) {
            v8bf t = *(const v8bf*)(inrow + c4 * 8);
#pragma unroll
            for (int e = 0; e < 8; ++e) BsT[c4 * 8 + e][threadIdx.x] = t[e];
        }
    }
    __syncthreads();

    v8f acc0 = {}, acc1 = {};
    const bf16* wrow = W + (size_t)(co0 + l16) * C_DIM + hf * 16;
    for (int k0 = 0; k0 < C_DIM; k0 += 32) {
        FragAB a;                 // A fragment: two contiguous 16B runs per lane
        a.p.lo = *(const v8bf*)(wrow + k0);
        a.p.hi = *(const v8bf*)(wrow + k0 + 8);
        FragAB b0, b1;            // B fragments: contiguous K in transposed LDS
        b0.p.lo = *(const v8bf*)&BsT[l16][k0 + hf * 16];
        b0.p.hi = *(const v8bf*)&BsT[l16][k0 + hf * 16 + 8];
        b1.p.lo = *(const v8bf*)&BsT[16 + l16][k0 + hf * 16];
        b1.p.hi = *(const v8bf*)&BsT[16 + l16][k0 + hf * 16 + 8];
        acc0 = __builtin_amdgcn_wmma_f32_16x16x32_bf16(false, a.v, false, b0.v,
                                                       (short)0, acc0, false, false);
        acc1 = __builtin_amdgcn_wmma_f32_16x16x32_bf16(false, a.v, false, b1.v,
                                                       (short)0, acc1, false, false);
    }

    // D layout: VGPR r, lane l -> M = hf*8 + r, N = l16 (frag0) / 16+l16 (frag1)
    int co_base = co0 + hf * 8;
    if (mode == 2) {
#pragma unroll
        for (int r = 0; r < 8; ++r) {
            int co = co_base + r;
            size_t i0 = ((size_t)b * C_DIM + co) * N_PIX + n0 + l16;
            size_t i1 = i0 + 16;
            outf[i0] = acc0[r] + bias[co] + resid[i0];
            outf[i1] = acc1[r] + bias[co] + resid[i1];
        }
    } else {
        const float sc = (mode == 3) ? 0.17677669529663687f : 1.0f;  // 32^-0.5
        int hh = co_base >> 5, dbase = co_base & 31;
        v8bf o0, o1;
#pragma unroll
        for (int r = 0; r < 8; ++r) {
            o0[r] = (bf16)((acc0[r] + bias[co_base + r]) * sc);
            o1[r] = (bf16)((acc1[r] + bias[co_base + r]) * sc);
        }
        size_t rowb = (size_t)(b * HEADS + hh) * N_PIX;
        *(v8bf*)&outb[(rowb + n0 + l16) * HD + dbase]      = o0;
        *(v8bf*)&outb[(rowb + n0 + 16 + l16) * HD + dbase] = o1;
    }
}

// ---------------------------------------------------------------------------
// Flash attention. Block = 4 waves sharing one (b,h); each wave owns a 16-row
// i-tile. k/v tiles (32 j) staged cooperatively in double-buffered LDS.
// Logical j <-> hardware-K bijection for the P.V^T WMMA is permuted:
//   sigma(j) = 2*(j&15) + (j>>4)
// so that the per-row (p0,p1) pair lands in ADJACENT physical columns of PS
// (one packed ds_store_b32), while Vs rows are staged at sigma(j).
// qT/kT/vT: [b][h][N][hd] bf16 ; out: [b][c][N] bf16.
// ---------------------------------------------------------------------------
__global__ void attn_kernel(const bf16* __restrict__ qT,
                            const bf16* __restrict__ kT,
                            const bf16* __restrict__ vT,
                            bf16* __restrict__ outb) {
    int wv   = threadIdx.x >> 5;             // 0..3
    int lane = threadIdx.x & 31;
    int bh   = blockIdx.x / 64;              // 64 blocks per (b,h)
    int i0   = (blockIdx.x % 64) * 64 + wv * 16;
    int hf   = lane >> 4;
    int l16  = lane & 15;

    const bf16* qp = qT + (size_t)bh * N_PIX * HD;
    const bf16* kp = kT + (size_t)bh * N_PIX * HD;
    const bf16* vp = vT + (size_t)bh * N_PIX * HD;

    // A fragment of q (scale already folded in projection)
    FragAB aq;
    {
        const bf16* qrow = qp + (size_t)(i0 + l16) * HD + hf * 16;
        aq.p.lo = *(const v8bf*)qrow;
        aq.p.hi = *(const v8bf*)(qrow + 8);
    }

    __shared__ bf16 Ks[2][32][40];           // [j][d], padded
    __shared__ bf16 Vs[2][32][40];           // [d][sigma(j)], padded
    __shared__ bf16 PS[4][16][40];           // per-wave P relayout, padded
    bf16 (*ps)[40] = PS[wv];

    int jrow  = threadIdx.x >> 2;            // 0..31 : staging row
    int chunk = threadIdx.x & 3;             // 0..3  : 8-elem chunk
    int sj    = 2 * (jrow & 15) + (jrow >> 4);   // sigma(jrow)

    float mrow[8], lrow[8];
#pragma unroll
    for (int r = 0; r < 8; ++r) { mrow[r] = -1e30f; lrow[r] = 0.f; }
    v8f O0 = {}, O1 = {};

    int db = 0;
    for (int j0 = 0; j0 < N_PIX; j0 += 32, db ^= 1) {
        // ---- prefetch next tile through L2 (global_prefetch_b8)
        if (j0 + 32 < N_PIX) {
            __builtin_prefetch(kp + (size_t)(j0 + 32 + jrow) * HD + chunk * 8, 0, 1);
            __builtin_prefetch(vp + (size_t)(j0 + 32 + jrow) * HD + chunk * 8, 0, 1);
        }
        // ---- cooperative staging (b128 reads; Vs transposed+permuted scatter)
        {
            v8bf kv = *(const v8bf*)(kp + (size_t)(j0 + jrow) * HD + chunk * 8);
            *(v8bf*)&Ks[db][jrow][chunk * 8] = kv;
            v8bf vv = *(const v8bf*)(vp + (size_t)(j0 + jrow) * HD + chunk * 8);
#pragma unroll
            for (int e = 0; e < 8; ++e) Vs[db][chunk * 8 + e][sj] = vv[e];
        }
        __syncthreads();

        // ---- S = q . k^T  (B frag: col j fixed per lane, K=d contiguous)
        FragAB bk0, bk1;
        bk0.p.lo = *(const v8bf*)&Ks[db][l16][hf * 16];
        bk0.p.hi = *(const v8bf*)&Ks[db][l16][hf * 16 + 8];
        bk1.p.lo = *(const v8bf*)&Ks[db][16 + l16][hf * 16];
        bk1.p.hi = *(const v8bf*)&Ks[db][16 + l16][hf * 16 + 8];
        v8f z = {};
        v8f S0 = __builtin_amdgcn_wmma_f32_16x16x32_bf16(false, aq.v, false, bk0.v,
                                                         (short)0, z, false, false);
        v8f S1 = __builtin_amdgcn_wmma_f32_16x16x32_bf16(false, aq.v, false, bk1.v,
                                                         (short)0, z, false, false);

        // ---- online softmax (row = hf*8 + r, its 16 columns live across lanes)
#pragma unroll
        for (int r = 0; r < 8; ++r) {
            float t = fmaxf(S0[r], S1[r]);
#pragma unroll
            for (int m = 1; m < 16; m <<= 1) t = fmaxf(t, __shfl_xor(t, m, 32));
            float mn    = fmaxf(mrow[r], t);
            float alpha = __expf(mrow[r] - mn);
            mrow[r] = mn;
            float p0 = __expf(S0[r] - mn);
            float p1 = __expf(S1[r] - mn);
            float srow = p0 + p1;
#pragma unroll
            for (int m = 1; m < 16; m <<= 1) srow += __shfl_xor(srow, m, 32);
            lrow[r] = lrow[r] * alpha + srow;
            O0[r] *= alpha;
            O1[r] *= alpha;
            // paired store: physical cols sigma(l16)=2*l16, sigma(16+l16)=2*l16+1
            v2bf pp;
            pp[0] = (bf16)p0;
            pp[1] = (bf16)p1;
            *(v2bf*)&ps[hf * 8 + r][2 * l16] = pp;
        }
        asm volatile("s_wait_dscnt 0x0" ::: "memory");

        // ---- O += P . v^T (both operands indexed by physical K = sigma(j))
        FragAB ap;
        ap.p.lo = *(const v8bf*)&ps[l16][hf * 16];
        ap.p.hi = *(const v8bf*)&ps[l16][hf * 16 + 8];
        FragAB bv0, bv1;
        bv0.p.lo = *(const v8bf*)&Vs[db][l16][hf * 16];
        bv0.p.hi = *(const v8bf*)&Vs[db][l16][hf * 16 + 8];
        bv1.p.lo = *(const v8bf*)&Vs[db][16 + l16][hf * 16];
        bv1.p.hi = *(const v8bf*)&Vs[db][16 + l16][hf * 16 + 8];
        O0 = __builtin_amdgcn_wmma_f32_16x16x32_bf16(false, ap.v, false, bv0.v,
                                                     (short)0, O0, false, false);
        O1 = __builtin_amdgcn_wmma_f32_16x16x32_bf16(false, ap.v, false, bv1.v,
                                                     (short)0, O1, false, false);
    }

    // ---- epilogue: normalize rows, vector store [b][h*32+d][i0+hf*8 .. +7]
    int b = bh >> 3, h = bh & 7;
    v8bf o0, o1;
#pragma unroll
    for (int r = 0; r < 8; ++r) {
        float inv = 1.0f / lrow[r];
        o0[r] = (bf16)(O0[r] * inv);
        o1[r] = (bf16)(O1[r] * inv);
    }
    size_t cbase = (size_t)(b * C_DIM + h * HD);
    *(v8bf*)&outb[(cbase + l16)      * N_PIX + i0 + hf * 8] = o0;
    *(v8bf*)&outb[(cbase + 16 + l16) * N_PIX + i0 + hf * 8] = o1;
}

// ---------------------------------------------------------------------------
extern "C" void kernel_launch(void* const* d_in, const int* in_sizes, int n_in,
                              void* d_out, int out_size, void* d_ws, size_t ws_size,
                              hipStream_t stream) {
    (void)in_sizes; (void)n_in; (void)out_size; (void)ws_size;
    const float* x      = (const float*)d_in[0];
    const float* norm_w = (const float*)d_in[1];
    const float* norm_b = (const float*)d_in[2];
    const float* q_w    = (const float*)d_in[3];
    const float* q_b    = (const float*)d_in[4];
    const float* k_w    = (const float*)d_in[5];
    const float* k_b    = (const float*)d_in[6];
    const float* v_w    = (const float*)d_in[7];
    const float* v_b    = (const float*)d_in[8];
    const float* p_w    = (const float*)d_in[9];
    const float* p_b    = (const float*)d_in[10];
    float* out = (float*)d_out;

    char* ws = (char*)d_ws;
    size_t off = 0;
    auto alloc = [&](size_t bytes) -> void* {
        void* p = ws + off;
        off += (bytes + 255) & ~(size_t)255;
        return p;
    };
    const size_t tensorB = (size_t)B_DIM * C_DIM * N_PIX * sizeof(bf16);   // 8 MB
    bf16* hbf  = (bf16*)alloc(tensorB);
    bf16* qTd  = (bf16*)alloc(tensorB);
    bf16* kTd  = (bf16*)alloc(tensorB);
    bf16* vTd  = (bf16*)alloc(tensorB);
    bf16* aOut = (bf16*)alloc(tensorB);
    bf16* wq   = (bf16*)alloc((size_t)C_DIM * C_DIM * sizeof(bf16));
    bf16* wk   = (bf16*)alloc((size_t)C_DIM * C_DIM * sizeof(bf16));
    bf16* wv   = (bf16*)alloc((size_t)C_DIM * C_DIM * sizeof(bf16));
    bf16* wp   = (bf16*)alloc((size_t)C_DIM * C_DIM * sizeof(bf16));

    const int WN = C_DIM * C_DIM;            // 65536
    cvt_w_kernel<<<WN / 256, 256, 0, stream>>>(q_w, wq, WN);
    cvt_w_kernel<<<WN / 256, 256, 0, stream>>>(k_w, wk, WN);
    cvt_w_kernel<<<WN / 256, 256, 0, stream>>>(v_w, wv, WN);
    cvt_w_kernel<<<WN / 256, 256, 0, stream>>>(p_w, wp, WN);

    groupnorm_kernel<<<B_DIM * GROUPS, 256, 0, stream>>>(x, norm_w, norm_b, hbf);

    const int gemmBlocks = B_DIM * 2 * (N_PIX / 32);   // 1024
    gemm_kernel<<<gemmBlocks, 256, 0, stream>>>(wq, hbf, q_b, qTd, nullptr, nullptr, 3);
    gemm_kernel<<<gemmBlocks, 256, 0, stream>>>(wk, hbf, k_b, kTd, nullptr, nullptr, 1);
    gemm_kernel<<<gemmBlocks, 256, 0, stream>>>(wv, hbf, v_b, vTd, nullptr, nullptr, 1);

    const int attnBlocks = B_DIM * HEADS * 64;          // 2048 (4 i-tiles/block)
    attn_kernel<<<attnBlocks, 128, 0, stream>>>(qTd, kTd, vTd, aOut);

    gemm_kernel<<<gemmBlocks, 256, 0, stream>>>(wp, aOut, p_b, nullptr, out, x, 2);
}
